// MoEUpdateMLP_43508018709224
// MI455X (gfx1250) — matrix operand order repaired and checked by reading, not compiled, not run on hardware
//
#include <hip/hip_runtime.h>
#include <hip/hip_bf16.h>

#define NEXP   8
#define RC     8
#define IN_C   192
#define HID    384
#define OUT_C  192
#define HW     16384          // 128*128
#define NPIX   65536          // 4*128*128
#define TILE_P 128
#define SA     392            // LDS activation row stride (halves), mult of 8, >=384

typedef _Float16 half_t;
typedef __attribute__((ext_vector_type(16))) _Float16 v16h;
typedef __attribute__((ext_vector_type(8)))  _Float16 v8h;
typedef __attribute__((ext_vector_type(2)))  _Float16 v2h;
typedef __attribute__((ext_vector_type(8)))  float    v8f;

// Branchless tanh-form GELU: gelu(v) ~= v * sigmoid(1.5957691*v + 0.0713548*v^3)
__device__ __forceinline__ float gelu_fast(float v) {
    float z = v * (1.5957691216057308f + 0.07135481283607f * v * v);
    return v * __builtin_amdgcn_rcpf(1.0f + __expf(-z));
}

// ---------------- router: per-pixel top-2 + gates, scatter into expert buckets ----------
__global__ void router_kernel(const float* __restrict__ ri,
                              const float* __restrict__ rW,
                              const float* __restrict__ rb,
                              int*   __restrict__ counts,
                              int*   __restrict__ perm,
                              float* __restrict__ pgate) {
    int pix = blockIdx.x * blockDim.x + threadIdx.x;
    if (pix >= NPIX) return;
    int b = pix >> 14;
    int s = pix & (HW - 1);
    float rc[RC];
#pragma unroll
    for (int c = 0; c < RC; ++c)
        rc[c] = ri[(size_t)b * RC * HW + (size_t)c * HW + s];
    float lg[NEXP];
#pragma unroll
    for (int e = 0; e < NEXP; ++e) {
        float acc = rb[e];
#pragma unroll
        for (int c = 0; c < RC; ++c) acc += rW[e * RC + c] * rc[c];
        lg[e] = acc;
    }
    int i0 = 0;
#pragma unroll
    for (int e = 1; e < NEXP; ++e) if (lg[e] > lg[i0]) i0 = e;   // first max (tie -> low idx)
    int i1 = (i0 == 0) ? 1 : 0;
#pragma unroll
    for (int e = 0; e < NEXP; ++e) if (e != i0 && lg[e] > lg[i1]) i1 = e;
    float m  = fmaxf(lg[i0], lg[i1]);
    float e0 = __expf(lg[i0] - m), e1 = __expf(lg[i1] - m);
    float inv = 1.0f / (e0 + e1);
    int s0 = atomicAdd(&counts[i0], 1);
    perm [i0 * NPIX + s0] = pix;
    pgate[i0 * NPIX + s0] = e0 * inv;
    int s1 = atomicAdd(&counts[i1], 1);
    perm [i1 * NPIX + s1] = pix;
    pgate[i1 * NPIX + s1] = e1 * inv;
}

// ---------------- f32 -> f16 weight conversion ----------------
__global__ void cvt_kernel(const float* __restrict__ src, half_t* __restrict__ dst, int n) {
    int i = blockIdx.x * blockDim.x + threadIdx.x;
    if (i < n) dst[i] = (half_t)src[i];
}

// ---------------- WMMA grouped GEMM (column-split across waves) ----------------
__device__ __forceinline__ v16h load_a_frag(const half_t* rowp, int khalf, int kt) {
    // 16-bit A 16x32: lane holds row M=lane%16; K contiguous [khalf*16, +16)
    const half_t* p = rowp + kt * 32 + khalf * 16;
    v8h lo = *(const v8h*)(p);
    v8h hi = *(const v8h*)(p + 8);
    v16h a;
#pragma unroll
    for (int j = 0; j < 8; ++j) { a[j] = lo[j]; a[8 + j] = hi[j]; }
    return a;
}

template <int KT, int NT, bool ACT, bool FINAL>
__device__ __forceinline__ void gemm_stage(const half_t* __restrict__ actIn,  // LDS, stride SA
                                           half_t* __restrict__ actOut,      // LDS (if !FINAL)
                                           const half_t* __restrict__ Wg,    // [NT*16][KT*32] row-major f16
                                           const float* __restrict__ bias,
                                           int wave, int lane,
                                           const int* plist, const float* gl, int rows,
                                           float* __restrict__ out) {
    const int khalf = lane >> 4;
    const int lan   = lane & 15;
    const int Kdim  = KT * 32;

    // Each wave owns column tiles nt = wave, wave+8, ... ; B-fragments are loaded
    // from global exactly once per block (no cross-wave redundancy), A streamed from LDS.
    for (int nt = wave; nt < NT; nt += 8) {
        const int ch = nt * 16 + lan;                            // B/C/D: N = lane%16
        const half_t* wp = Wg + (size_t)ch * Kdim + khalf * 16;  // 32 contiguous bytes / k-tile
        if (nt + 8 < NT) __builtin_prefetch(wp + (size_t)128 * Kdim, 0, 1);
        v16h bfrag[KT];
#pragma unroll
        for (int kt = 0; kt < KT; ++kt) bfrag[kt] = *(const v16h*)(wp + kt * 32);
        const float bv = bias[ch];

#pragma unroll 2
        for (int m = 0; m < 8; ++m) {                            // all 128 rows
            const half_t* rowp = actIn + (size_t)(m * 16 + lan) * SA;
            v8f acc = {};
#pragma unroll
            for (int kt = 0; kt < KT; ++kt) {
                v16h a = load_a_frag(rowp, khalf, kt);
                acc = __builtin_amdgcn_wmma_f32_16x16x32_f16(false, a, false, bfrag[kt],
                                                             (short)0, acc, false, false);
            }
            if (!FINAL) {
#pragma unroll
                for (int r = 0; r < 8; ++r) {                    // D: VGPR r -> row M = khalf*8+r
                    float v = acc[r] + bv;
                    if (ACT) v = gelu_fast(v);
                    int row = m * 16 + khalf * 8 + r;
                    actOut[(size_t)row * SA + ch] = (half_t)v;
                }
            } else {
#pragma unroll
                for (int r = 0; r < 8; ++r) {
                    int row = m * 16 + khalf * 8 + r;
                    if (row < rows) {
                        float v = (acc[r] + bv) * gl[row];
                        int pix = plist[row];
                        int b = pix >> 14, sp = pix & (HW - 1);
                        atomicAdd(&out[(size_t)b * OUT_C * HW + (size_t)ch * HW + sp], v);
                    }
                }
            }
        }
    }
}

__global__ __launch_bounds__(256)
void moe_mlp_kernel(const float* __restrict__ x,
                    const half_t* __restrict__ W1h, const float* __restrict__ b1,
                    const half_t* __restrict__ W2h, const float* __restrict__ b2,
                    const half_t* __restrict__ W3h, const float* __restrict__ b3,
                    const int* __restrict__ counts,
                    const int* __restrict__ perm, const float* __restrict__ pgate,
                    float* __restrict__ out) {
    extern __shared__ char smem[];
    int*    plist = (int*)smem;                    // 128 ints
    float*  gl    = (float*)(smem + 512);          // 128 floats
    half_t* actA  = (half_t*)(smem + 1024);        // 128*SA halves
    half_t* actB  = actA + (size_t)TILE_P * SA;    // 128*SA halves

    const int e   = blockIdx.y;
    const int cnt = counts[e];
    const int n0  = blockIdx.x * TILE_P;
    if (n0 >= cnt) return;                          // uniform exit
    const int rows = min(TILE_P, cnt - n0);
    const int tid  = threadIdx.x;
    const int wave = tid >> 5;
    const int lane = tid & 31;

    for (int i = tid; i < TILE_P; i += 256) {
        int pix = 0; float g = 0.f;
        if (i < rows) {
            pix = perm [(size_t)e * NPIX + n0 + i];
            g   = pgate[(size_t)e * NPIX + n0 + i];
        }
        plist[i] = pix; gl[i] = g;
    }
    __syncthreads();

    // gather x tile (pixel-major, f16) into LDS: 2 threads per pixel, 96 channels each.
    // Padded rows (p >= rows) read pixel 0 (plist zero-filled -> valid address) and write 0.
    {
        const int p      = tid >> 1;
        const int c0     = (tid & 1) * (IN_C / 2);
        const bool valid = (p < rows);
        const int pix    = plist[p];
        const int b      = pix >> 14;
        const int s      = pix & (HW - 1);
        const float* xp  = x + (size_t)b * (IN_C * HW) + (size_t)c0 * HW + s;
        v2h* dst = (v2h*)(actA + (size_t)p * SA + c0);
#pragma unroll 8
        for (int c = 0; c < IN_C / 2; c += 2) {
            float v0 = xp[(size_t)(c + 0) * HW];
            float v1 = xp[(size_t)(c + 1) * HW];
            v2h h;
            h[0] = valid ? (half_t)v0 : (half_t)0.f;
            h[1] = valid ? (half_t)v1 : (half_t)0.f;
            dst[c >> 1] = h;
        }
    }
    __syncthreads();

    const half_t* W1e = W1h + (size_t)e * HID * IN_C;
    const half_t* W2e = W2h + (size_t)e * HID * HID;
    const half_t* W3e = W3h + (size_t)e * OUT_C * HID;

    gemm_stage< 6, 24, true,  false>(actA, actB, W1e, b1 + e * HID,   wave, lane, nullptr, nullptr, 0, nullptr);
    __syncthreads();
    gemm_stage<12, 24, true,  false>(actB, actA, W2e, b2 + e * HID,   wave, lane, nullptr, nullptr, 0, nullptr);
    __syncthreads();
    gemm_stage<12, 12, false, true >(actA, nullptr, W3e, b3 + e * OUT_C, wave, lane, plist, gl, rows, out);
}

// ---------------- host ----------------
extern "C" void kernel_launch(void* const* d_in, const int* in_sizes, int n_in,
                              void* d_out, int out_size, void* d_ws, size_t ws_size,
                              hipStream_t stream) {
    const float* x  = (const float*)d_in[0];
    const float* ri = (const float*)d_in[1];
    const float* rW = (const float*)d_in[2];
    const float* rb = (const float*)d_in[3];
    const float* W1 = (const float*)d_in[4];
    const float* b1 = (const float*)d_in[5];
    const float* W2 = (const float*)d_in[6];
    const float* b2 = (const float*)d_in[7];
    const float* W3 = (const float*)d_in[8];
    const float* b3 = (const float*)d_in[9];
    float* out = (float*)d_out;

    char* ws = (char*)d_ws;
    size_t off = 0;
    auto walloc = [&](size_t bytes) -> void* {
        void* p = ws + off;
        off = (off + bytes + 255) & ~(size_t)255;
        return p;
    };
    int*    counts = (int*)   walloc(NEXP * sizeof(int));
    int*    perm   = (int*)   walloc((size_t)NEXP * NPIX * sizeof(int));
    float*  pgate  = (float*) walloc((size_t)NEXP * NPIX * sizeof(float));
    half_t* W1h    = (half_t*)walloc((size_t)NEXP * HID * IN_C * sizeof(half_t));
    half_t* W2h    = (half_t*)walloc((size_t)NEXP * HID * HID  * sizeof(half_t));
    half_t* W3h    = (half_t*)walloc((size_t)NEXP * OUT_C * HID * sizeof(half_t));

    hipMemsetAsync(counts, 0, NEXP * sizeof(int), stream);
    hipMemsetAsync(out, 0, (size_t)out_size * sizeof(float), stream);

    cvt_kernel<<<(NEXP * HID * IN_C  + 255) / 256, 256, 0, stream>>>(W1, W1h, NEXP * HID * IN_C);
    cvt_kernel<<<(NEXP * HID * HID   + 255) / 256, 256, 0, stream>>>(W2, W2h, NEXP * HID * HID);
    cvt_kernel<<<(NEXP * OUT_C * HID + 255) / 256, 256, 0, stream>>>(W3, W3h, NEXP * OUT_C * HID);

    router_kernel<<<NPIX / 256, 256, 0, stream>>>(ri, rW, rb, counts, perm, pgate);

    size_t shmem = 1024 + 2 * (size_t)TILE_P * SA * sizeof(half_t); // ~197 KB (WGP has 320 KB)
    dim3 grid(NPIX / TILE_P, NEXP);
    moe_mlp_kernel<<<grid, 256, shmem, stream>>>(x, W1h, b1, W2h, b2, W3h, b3,
                                                 counts, perm, pgate, out);
}
